// MultiheadAttention_4836133175765
// MI455X (gfx1250) — compile-verified, matrix operand order checked by reading, not run on hardware
//
#include <hip/hip_runtime.h>
#include <math.h>

// ---------------------------------------------------------------------------
// MultiheadAttention, restructured (softmax weights sum to 1, so the 1024x256
// value projection commutes past the attention contraction):
//   q = query @ Wq^T + bq                      (32,16,64)   -> kernel 1 (WMMA)
//   S = q @ key^T / 8 ; p = softmax_L(S)       (32,16,4096) -> kernel 2 (WMMA, split-L flash)
//   ctx = p @ value                            (32,16,256)  -> kernel 2 (WMMA)
//   out[b,h*64+d] = ctx[b,h,:]@Wv[h*64+d,:]+bv (32,1024)    -> kernel 3
// ~160MB unavoidable HBM traffic -> ~7us roofline; 512 WGs x 8 waves for
// enough memory-level parallelism to saturate 23.3 TB/s.
// ---------------------------------------------------------------------------

typedef _Float16 half_t;
typedef __attribute__((ext_vector_type(16))) _Float16 v16h;
typedef __attribute__((ext_vector_type(8)))  float    v8f;

#define B_DIM 32
#define L_DIM 4096
#define H_DIM 16
#define QS    64
#define VS    256
#define CHUNK 256
#define NCHUNK 16

__device__ __forceinline__ v8f wmma_f16(v16h a, v16h b, v8f c) {
  return __builtin_amdgcn_wmma_f32_16x16x32_f16(
      /*neg_a=*/false, a, /*neg_b=*/false, b,
      /*c_mod=*/(short)0, c, /*reuse_a=*/false, /*reuse_b=*/false);
}

// 16x32 fragment, "lane = row" layout (A-tiles, and B-tiles row-major along K).
// Element e of lane L: src[(r0+(L&15))*ld + k0 + 8*(L>>4) + 16*(e>>3) + (e&7)].
__device__ __forceinline__ v16h load_frag_rm(const float* __restrict__ base,
                                             int ld, int r0, int k0,
                                             float scale, int lane) {
  const int r = lane & 15, hf = lane >> 4;
  const float* p = base + (size_t)(r0 + r) * ld + k0 + hf * 8;
  v16h f;
#pragma unroll
  for (int e = 0; e < 16; ++e)
    f[e] = (half_t)(p[16 * (e >> 3) + (e & 7)] * scale);
  return f;
}

// B-tile (32x16 KxN) from source row-major along N: B[k,n] = src[(k0+k)*ld+n0+n].
__device__ __forceinline__ v16h load_frag_b_cm(const float* __restrict__ base,
                                               int ld, int k0, int n0, int lane) {
  const int n = lane & 15, hf = lane >> 4;
  const float* p = base + (size_t)(k0 + hf * 8) * ld + n0 + n;
  v16h f;
#pragma unroll
  for (int e = 0; e < 16; ++e)
    f[e] = (half_t)p[(size_t)(16 * (e >> 3) + (e & 7)) * ld];
  return f;
}

// A-tile (16x32) of f16 probabilities from LDS; row stride ldh halves.
__device__ __forceinline__ v16h load_frag_a_lds(const half_t* base, int ldh,
                                                int k0, int lane) {
  const int m = lane & 15, hf = lane >> 4;
  const half_t* p = base + m * ldh + k0 + hf * 8;
  v16h f;
#pragma unroll
  for (int e = 0; e < 16; ++e)
    f[e] = p[16 * (e >> 3) + (e & 7)];
  return f;
}

// ---------------------------------------------------------------------------
// Kernel 1: Q = query @ Wq^T + bq.  128 WMMA tiles (2 M x 64 N), 1 wave/tile.
// ---------------------------------------------------------------------------
__global__ __launch_bounds__(128) void k1_qproj(
    const float* __restrict__ query, const float* __restrict__ Wq,
    const float* __restrict__ bq, float* __restrict__ Q) {
  const int wid  = (blockIdx.x * blockDim.x + threadIdx.x) >> 5;  // 0..127
  const int lane = threadIdx.x & 31;
  const int mt = wid & 1;
  const int nt = wid >> 1;

  v8f c = {};
#pragma unroll 4
  for (int k0 = 0; k0 < 1024; k0 += 32) {
    v16h a = load_frag_rm(query, 1024, mt * 16, k0, 1.0f, lane);
    v16h b = load_frag_rm(Wq,    1024, nt * 16, k0, 1.0f, lane);
    c = wmma_f16(a, b, c);
  }
  const int n = lane & 15, hf = lane >> 4;
  const float bias = bq[nt * 16 + n];
#pragma unroll
  for (int v = 0; v < 8; ++v) {
    const int m = v + hf * 8;
    Q[(size_t)(mt * 16 + m) * 1024 + nt * 16 + n] = c[v] + bias;
  }
}

// ---------------------------------------------------------------------------
// Kernel 2: per (b, L-chunk of 256) partial flash attention.
// grid = 32*16 = 512 blocks, 256 threads (8 waves).  LDS ~25KB.
// ---------------------------------------------------------------------------
__global__ __launch_bounds__(256) void k2_attn_partial(
    const float* __restrict__ Q,      // (32, 1024) = (b, h*64+d)
    const float* __restrict__ key,    // (32, 4096, 64)
    const float* __restrict__ value,  // (32, 4096, 256)
    float* __restrict__ ctxp,         // (32, 16, 16, 256) unnormalized partial
    float* __restrict__ msum) {       // (32, 16, 16, 2)   {rowmax, sum exp}
  __shared__ float  S[16][CHUNK];        // 16 KB scores
  __shared__ half_t P[16][CHUNK + 8];    // +16B pad -> conflict-free b128 reads
  __shared__ float  red[16][2];

  const int b  = blockIdx.x >> 4;
  const int ch = blockIdx.x & 15;
  const int l0 = ch * CHUNK;
  const int tid = threadIdx.x, wave = tid >> 5, lane = tid & 31;

  // q fragments (16 heads x 64); 1/sqrt(64) folded into the f16 conversion
  const float* qb = Q + (size_t)b * 1024;          // row h, ld = 64
  const v16h a0 = load_frag_rm(qb, QS, 0, 0,  0.125f, lane);
  const v16h a1 = load_frag_rm(qb, QS, 0, 32, 0.125f, lane);

  // ---- phase 1: S(16 x 256) = q @ key_chunk^T -----------------------------
  const float* kb = key + ((size_t)b * L_DIM + l0) * QS;
#pragma unroll
  for (int t = 0; t < CHUNK / 16 / 8; ++t) {
    const int ln0 = (wave + t * 8) * 16;
    v16h b0 = load_frag_rm(kb, QS, ln0, 0,  1.0f, lane);
    v16h b1 = load_frag_rm(kb, QS, ln0, 32, 1.0f, lane);
    v8f c = {};
    c = wmma_f16(a0, b0, c);
    c = wmma_f16(a1, b1, c);
    const int n = lane & 15, hf = lane >> 4;
#pragma unroll
    for (int v = 0; v < 8; ++v) S[v + hf * 8][ln0 + n] = c[v];
  }
  __syncthreads();

  // ---- phase 2: rowwise max / exp / sum; P = exp(S - m) as f16 ------------
#pragma unroll
  for (int rr = 0; rr < 2; ++rr) {
    const int r = wave * 2 + rr;
    float vals[CHUNK / 32];
    float mx = -INFINITY;
#pragma unroll
    for (int i = 0; i < CHUNK / 32; ++i) {
      vals[i] = S[r][lane + 32 * i];
      mx = fmaxf(mx, vals[i]);
    }
#pragma unroll
    for (int off = 16; off; off >>= 1) mx = fmaxf(mx, __shfl_xor(mx, off, 32));
    float sum = 0.0f;
#pragma unroll
    for (int i = 0; i < CHUNK / 32; ++i) {
      const float e = __expf(vals[i] - mx);
      sum += e;
      P[r][lane + 32 * i] = (half_t)e;
    }
#pragma unroll
    for (int off = 16; off; off >>= 1) sum += __shfl_xor(sum, off, 32);
    if (lane == 0) { red[r][0] = mx; red[r][1] = sum; }
  }
  __syncthreads();

  // ---- phase 3: ctx(16 x 256) = P @ value_chunk ---------------------------
  const float* vb = value + ((size_t)b * L_DIM + l0) * VS;
  const int n0 = wave * 32;                 // each wave owns 2 N-tiles of e
  v8f c0 = {}, c1 = {};
#pragma unroll 2
  for (int k0 = 0; k0 < CHUNK; k0 += 32) {
    // prefetch next k-step's 32 value rows (this wave's 128B column span)
    if (k0 + 32 < CHUNK)
      __builtin_prefetch(vb + (size_t)(k0 + 32 + lane) * VS + n0, 0, 0);
    v16h pa  = load_frag_a_lds(&P[0][0], CHUNK + 8, k0, lane);
    v16h vb0 = load_frag_b_cm(vb, VS, k0, n0,      lane);
    v16h vb1 = load_frag_b_cm(vb, VS, k0, n0 + 16, lane);
    c0 = wmma_f16(pa, vb0, c0);
    c1 = wmma_f16(pa, vb1, c1);
  }

  float* op = ctxp + (((size_t)b * NCHUNK + ch) * 16) * VS;
  const int n = lane & 15, hf = lane >> 4;
#pragma unroll
  for (int v = 0; v < 8; ++v) {
    const int m = v + hf * 8;
    op[(size_t)m * VS + n0 + n]      = c0[v];
    op[(size_t)m * VS + n0 + 16 + n] = c1[v];
  }
  if (tid < 16) {
    msum[(((size_t)b * NCHUNK + ch) * 16 + tid) * 2 + 0] = red[tid][0];
    msum[(((size_t)b * NCHUNK + ch) * 16 + tid) * 2 + 1] = red[tid][1];
  }
}

// ---------------------------------------------------------------------------
// Kernel 3: merge chunk partials (online softmax) + output projection.
// grid = 32 blocks (one per batch), 256 threads.
// ---------------------------------------------------------------------------
__global__ __launch_bounds__(256) void k3_reduce_out(
    const float* __restrict__ ctxp, const float* __restrict__ msum,
    const float* __restrict__ Wv, const float* __restrict__ bv,
    float* __restrict__ out) {
  __shared__ float ctx[16][VS];
  __shared__ float scl[16][NCHUNK];

  const int b = blockIdx.x;
  const int tid = threadIdx.x, wave = tid >> 5, lane = tid & 31;

  if (tid < 16) {
    const int h = tid;
    float mp[NCHUNK], sp[NCHUNK], mx = -INFINITY;
#pragma unroll
    for (int p = 0; p < NCHUNK; ++p) {
      mp[p] = msum[(((size_t)b * NCHUNK + p) * 16 + h) * 2 + 0];
      sp[p] = msum[(((size_t)b * NCHUNK + p) * 16 + h) * 2 + 1];
      mx = fmaxf(mx, mp[p]);
    }
    float denom = 0.0f;
#pragma unroll
    for (int p = 0; p < NCHUNK; ++p) denom += sp[p] * __expf(mp[p] - mx);
    const float inv = 1.0f / denom;
#pragma unroll
    for (int p = 0; p < NCHUNK; ++p) scl[h][p] = __expf(mp[p] - mx) * inv;
  }
  __syncthreads();

  // ctx[h][e] = sum_p ctxp[b][p][h][e] * scl[h][p]   (coalesced over e)
  for (int i = tid; i < 16 * VS; i += 256) {
    const int h = i >> 8, e = i & (VS - 1);
    float acc = 0.0f;
#pragma unroll
    for (int p = 0; p < NCHUNK; ++p)
      acc += ctxp[(((size_t)b * NCHUNK + p) * 16 + h) * VS + e] * scl[h][p];
    ctx[h][e] = acc;
  }
  __syncthreads();

  // out[b, o] = ctx[h(o)] . Wv[o] + bv[o], lane-parallel over the 256-dot
  for (int o = wave; o < 1024; o += 8) {
    const int h = o >> 6;
    const float* wr = Wv + (size_t)o * VS;
    float acc = 0.0f;
#pragma unroll 4
    for (int e = lane; e < VS; e += 32) acc += ctx[h][e] * wr[e];
#pragma unroll
    for (int off = 16; off; off >>= 1) acc += __shfl_xor(acc, off, 32);
    if (lane == 0) out[(size_t)b * 1024 + o] = acc + bv[o];
  }
}

// ---------------------------------------------------------------------------
extern "C" void kernel_launch(void* const* d_in, const int* in_sizes, int n_in,
                              void* d_out, int out_size, void* d_ws, size_t ws_size,
                              hipStream_t stream) {
  const float* query = (const float*)d_in[0];  // (32, 1024)
  const float* key   = (const float*)d_in[1];  // (32, 4096, 64)
  const float* value = (const float*)d_in[2];  // (32, 4096, 256)
  const float* Wq    = (const float*)d_in[3];  // (1024, 1024)
  const float* bq    = (const float*)d_in[4];  // (1024)
  const float* Wv    = (const float*)d_in[5];  // (1024, 256)
  const float* bv    = (const float*)d_in[6];  // (1024)
  float* out = (float*)d_out;                  // (32, 1024)

  char* ws = (char*)d_ws;
  float* Q    = (float*)ws;                                        // 128 KB
  float* ctxp = (float*)(ws + 131072);                             // 8 MB
  float* msum = (float*)(ws + 131072 +
                         (size_t)B_DIM * NCHUNK * 16 * VS * 4);    // 64 KB

  k1_qproj<<<32, 128, 0, stream>>>(query, Wq, bq, Q);
  k2_attn_partial<<<B_DIM * NCHUNK, 256, 0, stream>>>(Q, key, value, ctxp, msum);
  k3_reduce_out<<<B_DIM, 256, 0, stream>>>(ctxp, msum, Wv, bv, out);
}
